// MultiHeadAttention_15015205666959
// MI455X (gfx1250) — matrix operand order repaired and checked by reading, not compiled, not run on hardware
//
#include <hip/hip_runtime.h>

// ---------------------------------------------------------------------------
// MI455X (gfx1250) multi-head attention, bf16 WMMA pipeline, wave32.
// ---------------------------------------------------------------------------

typedef __attribute__((ext_vector_type(16))) __bf16        v16bf;
typedef __attribute__((ext_vector_type(8)))  float         v8f;
typedef __attribute__((ext_vector_type(4)))  unsigned int  v4u;
typedef __attribute__((ext_vector_type(2)))  int           v2i;

#define DEV static __device__ __forceinline__

#if defined(__has_builtin)
#if __has_builtin(__builtin_amdgcn_global_load_async_to_lds_b64) && \
    __has_builtin(__builtin_amdgcn_s_wait_asynccnt)
#define HAS_ASYNC_LDS 1
#endif
#endif
#ifndef HAS_ASYNC_LDS
#define HAS_ASYNC_LDS 0
#endif

DEV unsigned short f2bf(float f) {
  unsigned int u = __builtin_bit_cast(unsigned int, f);
  u += 0x7FFFu + ((u >> 16) & 1u);           // round-to-nearest-even
  return (unsigned short)(u >> 16);
}

DEV v8f zero8() {
  v8f z;
#pragma unroll
  for (int i = 0; i < 8; ++i) z[i] = 0.f;
  return z;
}

union Frag32B { v16bf bf; v4u u[2]; };

// A operand, 16x32 bf16 (MxK), row-major source, element stride `stride`.
// Lane m = L&15; lane-half selects K {0..7,16..23} vs {8..15,24..31}.
DEV v16bf load_a_frag(const unsigned short* base, int stride) {
  int lane = threadIdx.x & 31;
  int m  = lane & 15;
  int ko = (lane >> 4) << 3;                 // 0 or 8
  Frag32B f;
  f.u[0] = *(const v4u*)(base + m * stride + ko);
  f.u[1] = *(const v4u*)(base + m * stride + ko + 16);
  return f.bf;
}

// B operand, 32x16 bf16 (KxN), sourced from B^T (N rows of K contiguous).
// Lane = column n (L&15); lane-half selects K {0..7,16..23} vs {8..15,24..31}.
DEV v16bf load_b_frag(const unsigned short* baseT, int stride) {
  int lane = threadIdx.x & 31;
  int n  = lane & 15;
  int ko = (lane >> 4) << 3;                 // 0 or 8
  const unsigned short* p = baseT + n * stride + ko;
  Frag32B f;
  f.u[0] = *(const v4u*)(p);
  f.u[1] = *(const v4u*)(p + 16);
  return f.bf;
}

DEV v8f wmma_bf16(v16bf a, v16bf b, v8f c) {
  return __builtin_amdgcn_wmma_f32_16x16x32_bf16(
      false, a, false, b, (short)0, c, false, false);
}

#if HAS_ASYNC_LDS
// Stage a 32x32 bf16 slab (rows row0.., cols k..k+31 of A[*,K]) into LDS.
// 256 threads x 8 bytes each, ASYNCcnt-tracked.
DEV void stage_slab_async(unsigned short* ldsbase, const unsigned short* A,
                          int row0, int K, int k) {
  int t = threadIdx.x;
  int r = t >> 3, c = (t & 7) << 2;
  __builtin_amdgcn_global_load_async_to_lds_b64(
      (__attribute__((address_space(1))) v2i*)(A + (row0 + r) * K + k + c),
      (__attribute__((address_space(3))) v2i*)(ldsbase + r * 32 + c),
      0, 0);
}
#endif

// ---------------------------------------------------------------------------
// Kernel 1: f32 -> bf16 (x activations), 4 elems/thread
// ---------------------------------------------------------------------------
__global__ __launch_bounds__(256) void cvt_x_kernel(
    const float* __restrict__ in, unsigned short* __restrict__ out, int n) {
  int i = (blockIdx.x * 256 + threadIdx.x) * 4;
  if (i >= n) return;
  float4 f = *(const float4*)(in + i);
  unsigned int lo = (unsigned int)f2bf(f.x) | ((unsigned int)f2bf(f.y) << 16);
  unsigned int hi = (unsigned int)f2bf(f.z) | ((unsigned int)f2bf(f.w) << 16);
  *(uint2*)(out + i) = make_uint2(lo, hi);
}

// ---------------------------------------------------------------------------
// Kernel 2: W [K x N] f32 -> W^T [N x K] bf16
// ---------------------------------------------------------------------------
__global__ __launch_bounds__(256) void cvt_wt_kernel(
    const float* __restrict__ W, unsigned short* __restrict__ Wt) {
  int idx = blockIdx.x * 256 + threadIdx.x;   // over 1024*1024 outputs
  int j = idx >> 10;                          // output row  (= W column)
  int i = idx & 1023;                         // output col  (= W row)
  Wt[idx] = f2bf(W[i * 1024 + j]);
}

// ---------------------------------------------------------------------------
// Kernel 3: bf16 GEMM  out = A[MxK] @ B[KxN]  (B given as B^T), bf16 out.
// Block: 32 rows x 512 cols; A slab (32x32) double-buffered in LDS via async
// copy; each wave computes 32x64 (8 accum frags, 8 WMMA per k-step).
// mode 0: out -> [B,H,S,64]; mode 1: out -> V^T [B,H,64,S].
// ---------------------------------------------------------------------------
__global__ __launch_bounds__(256) void gemm_qkv_kernel(
    const unsigned short* __restrict__ A,
    const unsigned short* __restrict__ Bt,
    unsigned short* __restrict__ out,
    int M, int N, int K, int mode) {
  __shared__ __align__(16) unsigned short abuf[2][32 * 32];
  int lwave = threadIdx.x >> 5;
  int lane  = threadIdx.x & 31;
  int nCB   = N >> 9;                         // 512-wide col blocks
  int rb    = blockIdx.x / nCB;
  int cb    = blockIdx.x - rb * nCB;
  int row0  = rb << 5;
  int col0  = (cb << 9) + (lwave << 6);
  if (row0 >= M) return;

  v8f acc[2][4];
#pragma unroll
  for (int t = 0; t < 2; ++t)
#pragma unroll
    for (int j = 0; j < 4; ++j) acc[t][j] = zero8();

#if HAS_ASYNC_LDS
  stage_slab_async(&abuf[0][0], A, row0, K, 0);
#endif

  for (int k = 0; k < K; k += 32) {
    int cur = (k >> 5) & 1;
#if HAS_ASYNC_LDS
    if (k + 32 < K) {
      stage_slab_async(&abuf[cur ^ 1][0], A, row0, K, k + 32);
      __builtin_amdgcn_s_wait_asynccnt(1);    // slab k complete (in-order)
    } else {
      __builtin_amdgcn_s_wait_asynccnt(0);
    }
    __syncthreads();
#else
    {
      int t = threadIdx.x;
      int r = t >> 3, c = (t & 7) << 2;
      *(uint2*)(&abuf[cur][r * 32 + c]) =
          *(const uint2*)(A + (row0 + r) * K + k + c);
      __syncthreads();
    }
#endif
    v16bf a0 = load_a_frag(&abuf[cur][0], 32);
    v16bf a1 = load_a_frag(&abuf[cur][16 * 32], 32);
#pragma unroll
    for (int j = 0; j < 4; ++j) {
      v16bf b = load_b_frag(Bt + (col0 + j * 16) * K + k, K);
      acc[0][j] = wmma_bf16(a0, b, acc[0][j]);
      acc[1][j] = wmma_bf16(a1, b, acc[1][j]);
    }
#if !HAS_ASYNC_LDS
    __syncthreads();
#endif
  }

  int hl = lane >> 4, n = lane & 15;
#pragma unroll
  for (int t = 0; t < 2; ++t)
#pragma unroll
    for (int j = 0; j < 4; ++j)
#pragma unroll
      for (int r = 0; r < 8; ++r) {
        int row = row0 + t * 16 + r + (hl << 3);
        int col = col0 + j * 16 + n;
        int bz = row >> 11, s = row & 2047;   // S = 2048
        int h  = col >> 6,  hd = col & 63;    // HD = 64
        unsigned short v = f2bf(acc[t][j][r]);
        if (mode == 0)
          out[(((bz * 16 + h) * 2048 + s) << 6) + hd] = v;
        else
          out[(((bz * 16 + h) * 64 + hd) << 11) + s] = v;
      }
}

// ---------------------------------------------------------------------------
// Kernel 4: causal flash attention. One wave per 32-query tile (2 row tiles).
// Q,K: [B,H,S,64] bf16.  Vt: [B,H,64,S] bf16.  ctx out: [B,S,1024] bf16.
// K/V fragments are loaded once per 32-key block and reused by both row tiles.
// ---------------------------------------------------------------------------
__global__ __launch_bounds__(256) void attn_kernel(
    const unsigned short* __restrict__ Q,
    const unsigned short* __restrict__ Km,
    const unsigned short* __restrict__ Vt,
    unsigned short* __restrict__ ctx) {
  __shared__ __align__(16) unsigned short pbuf[8][32 * 32];
  int lwave = threadIdx.x >> 5;
  int lane  = threadIdx.x & 31;
  int wid   = blockIdx.x * 8 + lwave;
  int qt = wid & 63;                          // S/32 = 64 query tiles
  int bh = wid >> 6;                          // b*16 + h
  int qbase = qt << 5;
  const unsigned short* qp = Q  + bh * 2048 * 64;
  const unsigned short* kp = Km + bh * 2048 * 64;
  const unsigned short* vp = Vt + bh * 64 * 2048;
  unsigned short* pl = pbuf[lwave];

  // Q tile 32x64 -> 2 row tiles x 2 K-chunks, kept resident
  v16bf qf[2][2];
#pragma unroll
  for (int t = 0; t < 2; ++t) {
    qf[t][0] = load_a_frag(qp + (qbase + t * 16) * 64, 64);
    qf[t][1] = load_a_frag(qp + (qbase + t * 16) * 64 + 32, 64);
  }

  v8f o[2][4];
#pragma unroll
  for (int t = 0; t < 2; ++t)
#pragma unroll
    for (int d = 0; d < 4; ++d) o[t][d] = zero8();
  float mrow[2][8], lrow[2][8];
#pragma unroll
  for (int t = 0; t < 2; ++t)
#pragma unroll
    for (int r = 0; r < 8; ++r) { mrow[t][r] = -1e30f; lrow[t][r] = 0.f; }

  int hl = lane >> 4, n = lane & 15;
  int nkb = (qbase >> 5) + 1;                 // causal bound, 32-key blocks

  for (int kb = 0; kb < nkb; ++kb) {
    int kbase = kb << 5;
    // K fragments for this 32-key block (shared by both row tiles)
    v16bf k00 = load_b_frag(kp + kbase * 64, 64);
    v16bf k01 = load_b_frag(kp + kbase * 64 + 32, 64);
    v16bf k10 = load_b_frag(kp + (kbase + 16) * 64, 64);
    v16bf k11 = load_b_frag(kp + (kbase + 16) * 64 + 32, 64);

#pragma unroll
    for (int t = 0; t < 2; ++t) {
      v8f s0 = wmma_bf16(qf[t][0], k00, zero8());
      s0     = wmma_bf16(qf[t][1], k01, s0);
      v8f s1 = wmma_bf16(qf[t][0], k10, zero8());
      s1     = wmma_bf16(qf[t][1], k11, s1);

      float p0[8], p1[8], mx[8];
#pragma unroll
      for (int r = 0; r < 8; ++r) {
        int qi = qbase + t * 16 + r + (hl << 3);
        float a = s0[r] * 0.125f;             // 1/sqrt(64)
        float b = s1[r] * 0.125f;
        a = (kbase + n      > qi) ? -1e30f : a;
        b = (kbase + 16 + n > qi) ? -1e30f : b;
        p0[r] = a; p1[r] = b;
        mx[r] = fmaxf(a, b);
      }
#pragma unroll
      for (int off = 8; off >= 1; off >>= 1)
#pragma unroll
        for (int r = 0; r < 8; ++r)
          mx[r] = fmaxf(mx[r], __shfl_xor(mx[r], off, 32));

      float alpha[8], rs[8];
#pragma unroll
      for (int r = 0; r < 8; ++r) {
        float mn = fmaxf(mrow[t][r], mx[r]);
        alpha[r]   = __expf(mrow[t][r] - mn);
        mrow[t][r] = mn;
        p0[r] = __expf(p0[r] - mn);
        p1[r] = __expf(p1[r] - mn);
        rs[r] = p0[r] + p1[r];
      }
#pragma unroll
      for (int off = 8; off >= 1; off >>= 1)
#pragma unroll
        for (int r = 0; r < 8; ++r)
          rs[r] += __shfl_xor(rs[r], off, 32);
#pragma unroll
      for (int r = 0; r < 8; ++r)
        lrow[t][r] = lrow[t][r] * alpha[r] + rs[r];

      // P (16x32) C-layout -> LDS row-major (rows t*16..t*16+15)
#pragma unroll
      for (int r = 0; r < 8; ++r) {
        int m = t * 16 + r + (hl << 3);
        pl[m * 32 + n]      = f2bf(p0[r]);
        pl[m * 32 + 16 + n] = f2bf(p1[r]);
      }
      // rescale running O
#pragma unroll
      for (int d = 0; d < 4; ++d)
#pragma unroll
        for (int r = 0; r < 8; ++r) o[t][d][r] *= alpha[r];
    }

    // context accumulation: reload P as A fragments, V frags shared by tiles
    v16bf pa0 = load_a_frag(pl, 32);
    v16bf pa1 = load_a_frag(pl + 16 * 32, 32);
#pragma unroll
    for (int d = 0; d < 4; ++d) {
      v16bf vb = load_b_frag(vp + (d * 16) * 2048 + kbase, 2048);
      o[0][d] = wmma_bf16(pa0, vb, o[0][d]);
      o[1][d] = wmma_bf16(pa1, vb, o[1][d]);
    }
  }

  // normalize and emit ctx tile (bf16, [B,S,1024])
  int bz = bh >> 4, h = bh & 15;
#pragma unroll
  for (int t = 0; t < 2; ++t) {
    float inv[8];
#pragma unroll
    for (int r = 0; r < 8; ++r) inv[r] = 1.f / lrow[t][r];
#pragma unroll
    for (int d = 0; d < 4; ++d)
#pragma unroll
      for (int r = 0; r < 8; ++r) {
        int qi  = qbase + t * 16 + r + (hl << 3);
        int col = h * 64 + d * 16 + n;
        ctx[(bz * 2048 + qi) * 1024 + col] = f2bf(o[t][d][r] * inv[r]);
      }
  }
}

// ---------------------------------------------------------------------------
// Kernel 5: out = ctx[Mx1024] @ Wp + bp  (Wp given as Wp^T bf16), f32 out.
// Same 32x512 block structure with async LDS staging of the A slab.
// ---------------------------------------------------------------------------
__global__ __launch_bounds__(256) void proj_kernel(
    const unsigned short* __restrict__ A,
    const unsigned short* __restrict__ Bt,
    const float* __restrict__ bias,
    float* __restrict__ out, int M) {
  const int K = 1024;
  __shared__ __align__(16) unsigned short abuf[2][32 * 32];
  int lwave = threadIdx.x >> 5;
  int lane  = threadIdx.x & 31;
  int rb    = blockIdx.x >> 1;
  int cb    = blockIdx.x & 1;
  int row0  = rb << 5;
  int col0  = (cb << 9) + (lwave << 6);
  if (row0 >= M) return;

  v8f acc[2][4];
#pragma unroll
  for (int t = 0; t < 2; ++t)
#pragma unroll
    for (int j = 0; j < 4; ++j) acc[t][j] = zero8();

#if HAS_ASYNC_LDS
  stage_slab_async(&abuf[0][0], A, row0, K, 0);
#endif

  for (int k = 0; k < K; k += 32) {
    int cur = (k >> 5) & 1;
#if HAS_ASYNC_LDS
    if (k + 32 < K) {
      stage_slab_async(&abuf[cur ^ 1][0], A, row0, K, k + 32);
      __builtin_amdgcn_s_wait_asynccnt(1);
    } else {
      __builtin_amdgcn_s_wait_asynccnt(0);
    }
    __syncthreads();
#else
    {
      int t = threadIdx.x;
      int r = t >> 3, c = (t & 7) << 2;
      *(uint2*)(&abuf[cur][r * 32 + c]) =
          *(const uint2*)(A + (row0 + r) * K + k + c);
      __syncthreads();
    }
#endif
    v16bf a0 = load_a_frag(&abuf[cur][0], 32);
    v16bf a1 = load_a_frag(&abuf[cur][16 * 32], 32);
#pragma unroll
    for (int j = 0; j < 4; ++j) {
      v16bf b = load_b_frag(Bt + (col0 + j * 16) * K + k, K);
      acc[0][j] = wmma_bf16(a0, b, acc[0][j]);
      acc[1][j] = wmma_bf16(a1, b, acc[1][j]);
    }
#if !HAS_ASYNC_LDS
    __syncthreads();
#endif
  }

  int hl = lane >> 4, n = lane & 15;
#pragma unroll
  for (int j = 0; j < 4; ++j) {
    float bn = bias[col0 + j * 16 + n];
#pragma unroll
    for (int t = 0; t < 2; ++t)
#pragma unroll
      for (int r = 0; r < 8; ++r) {
        int row = row0 + t * 16 + r + (hl << 3);
        out[row * 1024 + col0 + j * 16 + n] = acc[t][j][r] + bn;
      }
  }
}

// ---------------------------------------------------------------------------
extern "C" void kernel_launch(void* const* d_in, const int* in_sizes, int n_in,
                              void* d_out, int out_size, void* d_ws, size_t ws_size,
                              hipStream_t stream) {
  const float* x  = (const float*)d_in[0];
  const float* Wq = (const float*)d_in[1];
  const float* Wk = (const float*)d_in[2];
  const float* Wv = (const float*)d_in[3];
  const float* Wp = (const float*)d_in[4];
  const float* bp = (const float*)d_in[5];
  float* out = (float*)d_out;

  const int M = 4 * 2048;                    // B*S = 8192
  const int D = 1024;

  char* ws = (char*)d_ws;
  size_t off = 0;
  auto carve = [&](size_t bytes) {
    void* p = ws + off;
    off += (bytes + 255) & ~(size_t)255;
    return p;
  };
  unsigned short* xb   = (unsigned short*)carve((size_t)M * D * 2);   // 16 MB
  unsigned short* wqt  = (unsigned short*)carve((size_t)D * D * 2);   //  2 MB
  unsigned short* wkt  = (unsigned short*)carve((size_t)D * D * 2);
  unsigned short* wvt  = (unsigned short*)carve((size_t)D * D * 2);
  unsigned short* wpt  = (unsigned short*)carve((size_t)D * D * 2);
  unsigned short* q    = (unsigned short*)carve((size_t)M * D * 2);   // [B,H,S,64]
  unsigned short* kbuf = (unsigned short*)carve((size_t)M * D * 2);   // [B,H,S,64]
  unsigned short* vt   = (unsigned short*)carve((size_t)M * D * 2);   // [B,H,64,S]
  unsigned short* ctx  = (unsigned short*)carve((size_t)M * D * 2);   // [B,S,1024]

  // 1) conversions
  cvt_x_kernel<<<(M * D / 4 + 255) / 256, 256, 0, stream>>>(x, xb, M * D);
  cvt_wt_kernel<<<(D * D) / 256, 256, 0, stream>>>(Wq, wqt);
  cvt_wt_kernel<<<(D * D) / 256, 256, 0, stream>>>(Wk, wkt);
  cvt_wt_kernel<<<(D * D) / 256, 256, 0, stream>>>(Wv, wvt);
  cvt_wt_kernel<<<(D * D) / 256, 256, 0, stream>>>(Wp, wpt);

  // 2) QKV projections: (8192/32) * (1024/512) = 512 blocks of 8 waves
  dim3 gemmGrid(512);
  gemm_qkv_kernel<<<gemmGrid, 256, 0, stream>>>(xb, wqt, q,    M, D, D, 0);
  gemm_qkv_kernel<<<gemmGrid, 256, 0, stream>>>(xb, wkt, kbuf, M, D, D, 0);
  gemm_qkv_kernel<<<gemmGrid, 256, 0, stream>>>(xb, wvt, vt,   M, D, D, 1);

  // 3) attention: B*H*(S/32) = 4096 waves -> 512 blocks
  attn_kernel<<<dim3(512), 256, 0, stream>>>(q, kbuf, vt, ctx);

  // 4) output projection + bias
  proj_kernel<<<gemmGrid, 256, 0, stream>>>(ctx, wpt, bp, out, M);
}
